// FCOS_RT_64003602645491
// MI455X (gfx1250) — compile-verified, matrix operand order checked by reading, not run on hardware
//
#include <hip/hip_runtime.h>
#include <hip/hip_bf16.h>

typedef __attribute__((ext_vector_type(16))) _Float16 v16h;
typedef __attribute__((ext_vector_type(8)))  _Float16 v8h;
typedef __attribute__((ext_vector_type(4)))  _Float16 v4h;
typedef __attribute__((ext_vector_type(8)))  float    v8f;

#define D 256
#define NCLS 80
#define TOPK 1000
#define NTOT 3000

// ---------------------------------------------------------------------------
// Weight repack: [layer][oc][ic][ky][kx] f32  ->  [layer][tap][kstep][oc][32] f16
// matching the WMMA A-fragment layout (lanes 0-15: K 0-7 / 16-23, lanes 16-31:
// K 8-15 / 24-31 -> contiguous 32-half rows per oc).
// ---------------------------------------------------------------------------
__global__ void pack_weights(const float* __restrict__ w, _Float16* __restrict__ wp) {
    int t = blockIdx.x * blockDim.x + threadIdx.x;
    const int total = 4 * 9 * 8 * 256 * 32;
    if (t >= total) return;
    int icp   = t & 31;
    int oc    = (t >> 5) & 255;
    int ks    = (t >> 13) & 7;
    int tap   = (t >> 16) % 9;
    int layer = (t >> 16) / 9;
    int ic = ks * 32 + icp;
    size_t inIdx  = (((size_t)layer * 256 + oc) * 256 + ic) * 9 + tap;
    size_t outIdx = ((((size_t)layer * 9 + tap) * 8 + ks) * 256 + oc) * 32 + icp;
    wp[outIdx] = (_Float16)w[inIdx];
}

// NCHW f32 -> HWC f16
__global__ void nchw_to_hwc(const float* __restrict__ in, _Float16* __restrict__ out, int HW) {
    int i = blockIdx.x * blockDim.x + threadIdx.x;
    int total = HW * D;
    if (i >= total) return;
    int c = i / HW, pix = i % HW;               // coalesced read of NCHW
    out[(size_t)pix * D + c] = (_Float16)in[i];
}

// assemble a 16-half WMMA fragment from two contiguous 8-half (b128) loads
__device__ __forceinline__ v16h ldfrag(const _Float16* p) {
    v8h lo = *(const v8h*)p;
    v8h hi = *(const v8h*)(p + 16);
    v16h r;
#pragma unroll
    for (int q = 0; q < 8; ++q) { r[q] = lo[q]; r[q + 8] = hi[q]; }
    return r;
}

// ---------------------------------------------------------------------------
// 3x3 conv 256->256, bias+ReLU, implicit GEMM on v_wmma_f32_16x16x32_f16.
// Block: 256 threads = 8 waves; wave w owns oc [32w, 32w+32) x 32 pixels.
// Register blocking 2x2: per K-step, 2 A-frags + 2 B-frags -> 4 WMMAs
// (1 ds_load_b128 + 1 global_load_b128 per WMMA).
// B operand staged in LDS (3 x 34 x 256 f16 halo tile, 51 KB).
// ---------------------------------------------------------------------------
__global__ __launch_bounds__(256) void conv3x3_wmma(
    const _Float16* __restrict__ in, _Float16* __restrict__ out,
    const _Float16* __restrict__ wp, const float* __restrict__ bias,
    int H, int W) {
    __shared__ _Float16 tile[3 * 34 * 256];
    const int tid = threadIdx.x;
    const int xBase = blockIdx.x * 32;
    const int y = blockIdx.y;

    // stage halo tile (zero-padded), vectorized by 4 halfs
    for (int e = tid; e < (3 * 34 * 256) / 4; e += 256) {
        int ic4 = e & 63;
        int cc  = (e >> 6) % 34;
        int rr  = (e >> 6) / 34;
        int gy = y + rr - 1, gx = xBase + cc - 1;
        v4h v = {};
        if (gy >= 0 && gy < H && gx >= 0 && gx < W)
            v = *(const v4h*)(in + ((size_t)(gy * W + gx)) * D + ic4 * 4);
        *(v4h*)(&tile[(rr * 34 + cc) * D + ic4 * 4]) = v;
    }
    __syncthreads();

    const int wave = tid >> 5, lane = tid & 31;
    const int n   = lane & 15;             // pixel-within-16 / oc-row within tile
    const int sel = (lane & 16) ? 8 : 0;   // K-half select per WMMA lane layout
    const int ocBase = wave * 32;
    const int ocL = ocBase + n;            // A-matrix M index (tile m=0)

    v8f acc00 = {}, acc01 = {}, acc10 = {}, acc11 = {};
    for (int tap = 0; tap < 9; ++tap) {
        const int r = tap / 3, c = tap % 3;
        const _Float16* bsrc = &tile[(r * 34 + (n + c)) * D];
#pragma unroll
        for (int ks = 0; ks < 8; ++ks) {
            const _Float16* ap0 = wp + ((size_t)(tap * 8 + ks) * 256 + ocL) * 32 + sel;
            v16h a0 = ldfrag(ap0);                 // oc tile m=0
            v16h a1 = ldfrag(ap0 + 16 * 32);       // oc tile m=1 (+16 oc rows)
            const _Float16* bp0 = bsrc + ks * 32 + sel;
            v16h b0 = ldfrag(bp0);                 // pixels 0..15
            v16h b1 = ldfrag(bp0 + 16 * D);        // pixels 16..31
            acc00 = __builtin_amdgcn_wmma_f32_16x16x32_f16(false, a0, false, b0, (short)0, acc00, false, false);
            acc01 = __builtin_amdgcn_wmma_f32_16x16x32_f16(false, a0, false, b1, (short)0, acc01, false, false);
            acc10 = __builtin_amdgcn_wmma_f32_16x16x32_f16(false, a1, false, b0, (short)0, acc10, false, false);
            acc11 = __builtin_amdgcn_wmma_f32_16x16x32_f16(false, a1, false, b1, (short)0, acc11, false, false);
        }
    }

    // bias + ReLU + store (C/D layout: VGPR q -> M=q (+8 hi lanes), N=lane&15)
    const int mHi = (lane & 16) ? 8 : 0;
    auto emit = [&](const v8f& acc, int ocB, int px) {
        size_t pixOff = ((size_t)(y * W + px)) * D;
#pragma unroll
        for (int q = 0; q < 8; ++q) {
            int oc = ocB + q + mHi;
            float v = acc[q] + bias[oc];
            v = v > 0.f ? v : 0.f;
            out[pixOff + oc] = (_Float16)v;
        }
    };
    emit(acc00, ocBase +  0, xBase + n);
    emit(acc01, ocBase +  0, xBase + 16 + n);
    emit(acc10, ocBase + 16, xBase + n);
    emit(acc11, ocBase + 16, xBase + 16 + n);
}

// ---------------------------------------------------------------------------
// Fused 1x1 heads + FCOS decode. One block (128 thr) per pixel.
// ---------------------------------------------------------------------------
__global__ __launch_bounds__(128) void head_decode(
    const _Float16* __restrict__ cf, const _Float16* __restrict__ rf,
    const float* __restrict__ w_cls, const float* __restrict__ b_cls,
    const float* __restrict__ w_reg, const float* __restrict__ b_reg,
    const float* __restrict__ w_ctn, const float* __restrict__ b_ctn,
    int W, float stride,
    float* __restrict__ boxes, float* __restrict__ scores, float* __restrict__ labels) {
    __shared__ float scf[D], srf[D];
    __shared__ float logits[NCLS];
    __shared__ float sreg[4];
    __shared__ float sctn;
    const int pix = blockIdx.x;
    const int t = threadIdx.x;
    for (int i = t; i < D; i += 128) {
        scf[i] = (float)cf[(size_t)pix * D + i];
        srf[i] = (float)rf[(size_t)pix * D + i];
    }
    __syncthreads();
    if (t < NCLS) {
        float s = b_cls[t];
        for (int i = 0; i < D; ++i) s += w_cls[t * D + i] * scf[i];
        logits[t] = s;
    } else if (t < NCLS + 4) {
        int r = t - NCLS;
        float s = b_reg[r];
        for (int i = 0; i < D; ++i) s += w_reg[r * D + i] * srf[i];
        sreg[r] = s;
    } else if (t == NCLS + 4) {
        float s = b_ctn[0];
        for (int i = 0; i < D; ++i) s += w_ctn[i] * srf[i];
        sctn = s;
    }
    __syncthreads();
    if (t == 0) {
        float ctnSig = 1.f / (1.f + __expf(-sctn));
        float best = -1.f; int bi = 0;
        for (int c = 0; c < NCLS; ++c) {
            float sc = sqrtf((1.f / (1.f + __expf(-logits[c]))) * ctnSig);
            if (sc > best) { best = sc; bi = c; }
        }
        int px = pix % W, py = pix / W;
        float ax = px + 0.5f, ay = py + 0.5f;
        float sc_ = stride / 1024.f;
        float x1 = (ax - __expf(sreg[0])) * sc_;
        float y1 = (ay - __expf(sreg[1])) * sc_;
        float x2 = (ax + __expf(sreg[2])) * sc_;
        float y2 = (ay + __expf(sreg[3])) * sc_;
        x1 = fminf(fmaxf(x1, 0.f), 1.f); y1 = fminf(fmaxf(y1, 0.f), 1.f);
        x2 = fminf(fmaxf(x2, 0.f), 1.f); y2 = fminf(fmaxf(y2, 0.f), 1.f);
        boxes[pix * 4 + 0] = x1; boxes[pix * 4 + 1] = y1;
        boxes[pix * 4 + 2] = x2; boxes[pix * 4 + 3] = y2;
        scores[pix] = best;
        labels[pix] = (float)bi;
    }
}

// ---------------------------------------------------------------------------
// Sorting (global bitonic, descending) + top-k gather + NMS
// ---------------------------------------------------------------------------
__global__ void sort_init(float* key, int* val, const float* scores, int n, int m) {
    int i = blockIdx.x * blockDim.x + threadIdx.x;
    if (i >= n) return;
    key[i] = (i < m) ? scores[i] : -1e30f;
    val[i] = i;
}

__global__ void bitonic_step(float* key, int* val, int n, int k, int j) {
    int i = blockIdx.x * blockDim.x + threadIdx.x;
    if (i >= n) return;
    int ixj = i ^ j;
    if (ixj > i) {
        float ki = key[i], kj = key[ixj];
        bool descBlock = ((i & k) == 0);
        bool doSwap = descBlock ? (ki < kj) : (ki > kj);
        if (doSwap) {
            key[i] = kj; key[ixj] = ki;
            int vi = val[i]; val[i] = val[ixj]; val[ixj] = vi;
        }
    }
}

__global__ void topk_gather(const float* key, const int* val,
                            const float* boxes, const float* labels,
                            float* outBoxes, float* outScores, float* outLabels,
                            int k, int outOff) {
    int i = blockIdx.x * blockDim.x + threadIdx.x;
    if (i >= k) return;
    int idx = val[i];
    outScores[outOff + i] = key[i];
    outLabels[outOff + i] = labels[idx];
    for (int q = 0; q < 4; ++q)
        outBoxes[(size_t)(outOff + i) * 4 + q] = boxes[(size_t)idx * 4 + q];
}

__global__ __launch_bounds__(1024) void nms_suppress(
    const int* __restrict__ order, const float* __restrict__ boxes,
    const float* __restrict__ scores, const float* __restrict__ labels,
    float* __restrict__ keepOut, int m) {
    __shared__ int keep[NTOT];
    int t = threadIdx.x;
    for (int i = t; i < m; i += 1024) {
        int oi = order[i];
        keep[i] = (scores[oi] >= 0.05f) ? 1 : 0;
    }
    __syncthreads();
    for (int i = 0; i < m; ++i) {
        if (keep[i]) {                    // uniform branch (LDS value)
            int oi = order[i];
            float x1i = boxes[oi * 4 + 0], y1i = boxes[oi * 4 + 1];
            float x2i = boxes[oi * 4 + 2], y2i = boxes[oi * 4 + 3];
            float li = labels[oi];
            float areai = (x2i - x1i) * (y2i - y1i);
            for (int j = i + 1 + t; j < m; j += 1024) {
                if (!keep[j]) continue;
                int oj = order[j];
                if (labels[oj] != li) continue;
                float bx1 = boxes[oj * 4 + 0], by1 = boxes[oj * 4 + 1];
                float bx2 = boxes[oj * 4 + 2], by2 = boxes[oj * 4 + 3];
                float xx1 = fmaxf(x1i, bx1), yy1 = fmaxf(y1i, by1);
                float xx2 = fminf(x2i, bx2), yy2 = fminf(y2i, by2);
                float inter = fmaxf(1e-28f, xx2 - xx1) * fmaxf(1e-28f, yy2 - yy1);
                float areaj = (bx2 - bx1) * (by2 - by1);
                float iou = inter / (areai + areaj - inter + 1e-14f);
                if (iou > 0.6f) keep[j] = 0;
            }
        }
        __syncthreads();
    }
    for (int i = t; i < m; i += 1024) keepOut[order[i]] = (float)keep[i];
}

// ---------------------------------------------------------------------------
// Host orchestration
// ---------------------------------------------------------------------------
extern "C" void kernel_launch(void* const* d_in, const int* in_sizes, int n_in,
                              void* d_out, int out_size, void* d_ws, size_t ws_size,
                              hipStream_t stream) {
    (void)in_sizes; (void)n_in; (void)out_size; (void)ws_size;
    const float* p3    = (const float*)d_in[0];
    const float* p4    = (const float*)d_in[1];
    const float* p5    = (const float*)d_in[2];
    const float* cls_w = (const float*)d_in[3];
    const float* cls_b = (const float*)d_in[4];
    const float* reg_w = (const float*)d_in[5];
    const float* reg_b = (const float*)d_in[6];
    const float* w_cls = (const float*)d_in[7];
    const float* b_cls = (const float*)d_in[8];
    const float* w_reg = (const float*)d_in[9];
    const float* b_reg = (const float*)d_in[10];
    const float* w_ctn = (const float*)d_in[11];
    const float* b_ctn = (const float*)d_in[12];

    // workspace carve-out
    size_t off = 0;
    auto carve = [&](size_t bytes) -> char* {
        off = (off + 255) & ~(size_t)255;
        char* p = (char*)d_ws + off;
        off += bytes;
        return p;
    };
    const size_t LAYER_HALFS = (size_t)9 * 8 * 256 * 32;     // 589824
    _Float16* wpc = (_Float16*)carve(4 * LAYER_HALFS * 2);
    _Float16* wpr = (_Float16*)carve(4 * LAYER_HALFS * 2);
    const size_t ACT = (size_t)128 * 128 * D * 2;            // 8 MB
    _Float16* X  = (_Float16*)carve(ACT);
    _Float16* T1 = (_Float16*)carve(ACT);
    _Float16* CF = (_Float16*)carve(ACT);
    _Float16* RF = (_Float16*)carve(ACT);
    float* lvlBoxes  = (float*)carve(16384 * 4 * sizeof(float));
    float* lvlScore  = (float*)carve(16384 * sizeof(float));
    float* lvlLabel  = (float*)carve(16384 * sizeof(float));
    float* sKey      = (float*)carve(16384 * sizeof(float));
    int*   sVal      = (int*)  carve(16384 * sizeof(int));
    float* nKey      = (float*)carve(4096 * sizeof(float));
    int*   nVal      = (int*)  carve(4096 * sizeof(int));

    float* outBoxes  = (float*)d_out;              // [3000][4]
    float* outScores = outBoxes + NTOT * 4;        // [3000]
    float* outLabels = outScores + NTOT;           // [3000]
    float* outKeep   = outLabels + NTOT;           // [3000]

    // repack tower weights into WMMA A-fragment layout
    {
        const int total = 4 * 9 * 8 * 256 * 32;
        dim3 g((total + 255) / 256);
        pack_weights<<<g, 256, 0, stream>>>(cls_w, wpc);
        pack_weights<<<g, 256, 0, stream>>>(reg_w, wpr);
    }

    auto run_sort = [&](float* key, int* val, int n) {
        dim3 g((n + 255) / 256);
        for (int k = 2; k <= n; k <<= 1)
            for (int j = k >> 1; j > 0; j >>= 1)
                bitonic_step<<<g, 256, 0, stream>>>(key, val, n, k, j);
    };

    const float* feats[3] = {p3, p4, p5};
    for (int l = 0; l < 3; ++l) {
        const int Hh = 128 >> l, Ww = 128 >> l;
        const int HW = Hh * Ww;
        const float strd = (float)(8 << l);

        nchw_to_hwc<<<dim3((HW * D + 255) / 256), 256, 0, stream>>>(feats[l], X, HW);

        dim3 cg(Ww / 32, Hh);
        // cls tower: X -> T1 -> CF -> T1 -> CF
        conv3x3_wmma<<<cg, 256, 0, stream>>>(X,  T1, wpc + 0 * LAYER_HALFS, cls_b + 0 * D, Hh, Ww);
        conv3x3_wmma<<<cg, 256, 0, stream>>>(T1, CF, wpc + 1 * LAYER_HALFS, cls_b + 1 * D, Hh, Ww);
        conv3x3_wmma<<<cg, 256, 0, stream>>>(CF, T1, wpc + 2 * LAYER_HALFS, cls_b + 2 * D, Hh, Ww);
        conv3x3_wmma<<<cg, 256, 0, stream>>>(T1, CF, wpc + 3 * LAYER_HALFS, cls_b + 3 * D, Hh, Ww);
        // reg tower: X -> T1 -> RF -> T1 -> RF
        conv3x3_wmma<<<cg, 256, 0, stream>>>(X,  T1, wpr + 0 * LAYER_HALFS, reg_b + 0 * D, Hh, Ww);
        conv3x3_wmma<<<cg, 256, 0, stream>>>(T1, RF, wpr + 1 * LAYER_HALFS, reg_b + 1 * D, Hh, Ww);
        conv3x3_wmma<<<cg, 256, 0, stream>>>(RF, T1, wpr + 2 * LAYER_HALFS, reg_b + 2 * D, Hh, Ww);
        conv3x3_wmma<<<cg, 256, 0, stream>>>(T1, RF, wpr + 3 * LAYER_HALFS, reg_b + 3 * D, Hh, Ww);

        head_decode<<<dim3(HW), 128, 0, stream>>>(CF, RF, w_cls, b_cls, w_reg, b_reg,
                                                  w_ctn, b_ctn, Ww, strd,
                                                  lvlBoxes, lvlScore, lvlLabel);

        sort_init<<<dim3((HW + 255) / 256), 256, 0, stream>>>(sKey, sVal, lvlScore, HW, HW);
        run_sort(sKey, sVal, HW);
        topk_gather<<<dim3((TOPK + 255) / 256), 256, 0, stream>>>(
            sKey, sVal, lvlBoxes, lvlLabel, outBoxes, outScores, outLabels, TOPK, l * TOPK);
    }

    // NMS over the 3000 concatenated detections
    sort_init<<<dim3((4096 + 255) / 256), 256, 0, stream>>>(nKey, nVal, outScores, 4096, NTOT);
    run_sort(nKey, nVal, 4096);
    nms_suppress<<<dim3(1), 1024, 0, stream>>>(nVal, outBoxes, outScores, outLabels, outKeep, NTOT);
}
